// APPNPNet_83133386981988
// MI455X (gfx1250) — matrix operand order repaired and checked by reading, not compiled, not run on hardware
//
#include <hip/hip_runtime.h>

typedef __attribute__((ext_vector_type(16))) __bf16 v16bf;
typedef __attribute__((ext_vector_type(8)))  __bf16 v8bf;
typedef __attribute__((ext_vector_type(8)))  float  v8f;

#define ALPHA_F 0.1f
#define KEEP_F  0.9f

// ---------------- utility kernels ----------------

__global__ void zero_f32(float* __restrict__ p, long long n) {
  long long i = (long long)blockIdx.x * blockDim.x + threadIdx.x;
  if (i < n) p[i] = 0.0f;
}

// W [K][Ncol] fp32  ->  WT [Ncol][K] bf16 (stored as ushort bits)
__global__ void cvt_transpose_bf16(const float* __restrict__ W,
                                   unsigned short* __restrict__ WT,
                                   int K, int Ncol) {
  long long i = (long long)blockIdx.x * blockDim.x + threadIdx.x;
  if (i >= (long long)K * Ncol) return;
  int k = (int)(i % K);
  int n = (int)(i / K);
  __bf16 b = (__bf16)W[(long long)k * Ncol + n];
  WT[(long long)n * K + k] = __builtin_bit_cast(unsigned short, b);
}

__global__ void deg_accum(const int* __restrict__ dst, float* __restrict__ deg, int E) {
  int i = blockIdx.x * blockDim.x + threadIdx.x;
  if (i < E) atomicAdd(&deg[dst[i]], 1.0f);
}

__global__ void deg_finalize(float* __restrict__ dinv, int N) {
  int i = blockIdx.x * blockDim.x + threadIdx.x;
  if (i < N) dinv[i] = rsqrtf(dinv[i] + 1.0f);  // +1 = self loop; deg >= 1 always
}

__global__ void edge_norm(const int* __restrict__ src, const int* __restrict__ dst,
                          const float* __restrict__ dinv, float* __restrict__ norm, int E) {
  int i = blockIdx.x * blockDim.x + threadIdx.x;
  if (i < E) norm[i] = KEEP_F * dinv[src[i]] * dinv[dst[i]];  // fold (1-alpha)
}

// ---------------- fused MLP via WMMA (bf16) ----------------
// One workgroup (128 thr = 4 waves) handles 16 rows of x.
// Stage 1: hmid[16][256] = relu(x[16][512] @ W1 + b1)   (each wave: 4 N-tiles)
// Stage 2: out[16][64]   = hmid @ W2 + b2               (each wave: 1 N-tile)
__global__ __launch_bounds__(128) void mlp_wmma(
    const float* __restrict__ x,
    const unsigned short* __restrict__ W1T,   // [256][512] bf16
    const float* __restrict__ b1,
    const unsigned short* __restrict__ W2T,   // [64][256] bf16
    const float* __restrict__ b2,
    float* __restrict__ h0,
    float* __restrict__ hA) {
  __shared__ __align__(16) __bf16 xt[16][528];   // 512 + 16 pad (16B-multiple stride)
  __shared__ __align__(16) __bf16 hm[16][272];   // 256 + 16 pad

  const int tid  = threadIdx.x;
  const int lane = tid & 31;
  const int wid  = tid >> 5;
  const long long row0 = (long long)blockIdx.x * 16;

  // ---- load & convert x tile (16 x 512 fp32 -> bf16 in LDS), coalesced float4 ----
  for (int f = tid; f < 16 * 128; f += 128) {
    const int r  = f >> 7;
    const int c4 = (f & 127) << 2;
    const float4 v = *(const float4*)(x + (row0 + r) * 512 + c4);
    xt[r][c4 + 0] = (__bf16)v.x;  xt[r][c4 + 1] = (__bf16)v.y;
    xt[r][c4 + 2] = (__bf16)v.z;  xt[r][c4 + 3] = (__bf16)v.w;
  }
  __syncthreads();

  const int rA   = lane & 15;       // A-matrix row / B-matrix column-in-tile
  const int hsel = lane >> 4;       // lane half
  const int c0   = hsel * 8;        // A frag K sub-offset  (K0-7 / K8-15)
  const int kh   = hsel * 8;        // B frag K sub-offset  (K0-7 / K8-15)

  // ---- stage 1 ----
  v8f acc[4] = {};
  for (int kb = 0; kb < 512; kb += 32) {
    union { v16bf v; v8bf h[2]; } a;
    a.h[0] = *(const v8bf*)&xt[rA][kb + c0];        // K = c0..c0+7
    a.h[1] = *(const v8bf*)&xt[rA][kb + 16 + c0];   // K = 16+c0..16+c0+7
#pragma unroll
    for (int j = 0; j < 4; ++j) {
      const int n = wid * 64 + j * 16 + rA;
      union { v16bf v; v8bf h[2]; } b;
      const unsigned short* bp = W1T + (long long)n * 512 + kb + kh;
      b.h[0] = *(const v8bf*)(bp);
      b.h[1] = *(const v8bf*)(bp + 16);
      acc[j] = __builtin_amdgcn_wmma_f32_16x16x32_bf16(
                   false, a.v, false, b.v, (short)0, acc[j], false, false);
    }
  }
  // bias + relu -> LDS bf16   (C layout: VGPR r -> M = r + 8*hsel, N = lane&15 tile col)
#pragma unroll
  for (int j = 0; j < 4; ++j) {
    const int n = wid * 64 + j * 16 + rA;
    const float bias = b1[n];
#pragma unroll
    for (int r = 0; r < 8; ++r) {
      float v = acc[j][r] + bias;
      v = v > 0.0f ? v : 0.0f;
      hm[r + 8 * hsel][n] = (__bf16)v;
    }
  }
  __syncthreads();

  // ---- stage 2 ----
  v8f acc2 = {};
  const int n2 = wid * 16 + rA;
  for (int kb = 0; kb < 256; kb += 32) {
    union { v16bf v; v8bf h[2]; } a;
    a.h[0] = *(const v8bf*)&hm[rA][kb + c0];
    a.h[1] = *(const v8bf*)&hm[rA][kb + 16 + c0];
    union { v16bf v; v8bf h[2]; } b;
    const unsigned short* bp = W2T + (long long)n2 * 256 + kb + kh;
    b.h[0] = *(const v8bf*)(bp);
    b.h[1] = *(const v8bf*)(bp + 16);
    acc2 = __builtin_amdgcn_wmma_f32_16x16x32_bf16(
               false, a.v, false, b.v, (short)0, acc2, false, false);
  }
  const float bias2 = b2[n2];
#pragma unroll
  for (int r = 0; r < 8; ++r) {
    const long long g = row0 + r + 8 * hsel;
    const float v = acc2[r] + bias2;
    h0[g * 64 + n2] = v;   // teleport term
    hA[g * 64 + n2] = v;   // initial h
  }
}

// ---------------- APPNP propagation ----------------

// hnxt = keep*dinv^2*hcur (self loop) + alpha*h0   -- full init, then edges atomically add
__global__ void prop_init(const float* __restrict__ hcur, const float* __restrict__ h0,
                          const float* __restrict__ dinv, float* __restrict__ hnxt,
                          long long nvec4) {
  long long i = (long long)blockIdx.x * blockDim.x + threadIdx.x;  // over N*16 float4
  if (i >= nvec4) return;
  const int row = (int)(i >> 4);
  float c = dinv[row];
  c = KEEP_F * c * c;
  const float4 h = ((const float4*)hcur)[i];
  const float4 z = ((const float4*)h0)[i];
  float4 o;
  o.x = c * h.x + ALPHA_F * z.x;
  o.y = c * h.y + ALPHA_F * z.y;
  o.z = c * h.z + ALPHA_F * z.z;
  o.w = c * h.w + ALPHA_F * z.w;
  ((float4*)hnxt)[i] = o;
}

// one wave32 per edge: 64 columns, 2 f32 per lane; norm already carries (1-alpha)
__global__ void prop_edges(const int* __restrict__ src, const int* __restrict__ dst,
                           const float* __restrict__ norm, const float* __restrict__ hcur,
                           float* __restrict__ hnxt, int E) {
  const int lane = threadIdx.x & 31;
  long long w = ((long long)blockIdx.x * blockDim.x + threadIdx.x) >> 5;
  const long long nw = ((long long)gridDim.x * blockDim.x) >> 5;
  for (long long e = w; e < E; e += nw) {
    const int s = src[e];
    const int d = dst[e];
    const float wt = norm[e];
    const float2 v = ((const float2*)(hcur + (long long)s * 64))[lane];
    float* o = hnxt + (long long)d * 64 + lane * 2;
    atomicAdd(o,     wt * v.x);
    atomicAdd(o + 1, wt * v.y);
  }
}

// ---------------- log-softmax over 64 classes (1 wave per row) ----------------
__global__ void log_softmax64(const float* __restrict__ h, float* __restrict__ out, int N) {
  const int lane = threadIdx.x & 31;
  const int wid  = threadIdx.x >> 5;
  const long long row = (long long)blockIdx.x * 8 + wid;
  if (row >= N) return;
  const float2 v = ((const float2*)(h + row * 64))[lane];
  float m = fmaxf(v.x, v.y);
#pragma unroll
  for (int off = 16; off > 0; off >>= 1) m = fmaxf(m, __shfl_xor(m, off));
  float s = __expf(v.x - m) + __expf(v.y - m);
#pragma unroll
  for (int off = 16; off > 0; off >>= 1) s += __shfl_xor(s, off);
  const float lse = m + __logf(s);
  float2 o;
  o.x = v.x - lse;
  o.y = v.y - lse;
  ((float2*)(out + row * 64))[lane] = o;
}

// ---------------- launcher ----------------

extern "C" void kernel_launch(void* const* d_in, const int* in_sizes, int n_in,
                              void* d_out, int out_size, void* d_ws, size_t ws_size,
                              hipStream_t stream) {
  const float* x  = (const float*)d_in[0];
  const int*   ei = (const int*)  d_in[1];   // edge_index flattened [2][E]
  const float* W1 = (const float*)d_in[2];
  const float* b1 = (const float*)d_in[3];
  const float* W2 = (const float*)d_in[4];
  const float* b2 = (const float*)d_in[5];

  const int N = in_sizes[0] / 512;   // 100000
  const int E = in_sizes[1] / 2;     // 3200000
  const int* src = ei;
  const int* dst = ei + E;

  char* wptr = (char*)d_ws;
  auto alloc = [&](size_t bytes) -> char* {
    char* p = wptr;
    wptr += (bytes + 255) & ~(size_t)255;
    return p;
  };
  unsigned short* W1T = (unsigned short*)alloc((size_t)256 * 512 * 2);
  unsigned short* W2T = (unsigned short*)alloc((size_t)64 * 256 * 2);
  float* dinv = (float*)alloc((size_t)N * 4);
  float* norm = (float*)alloc((size_t)E * 4);
  float* h0   = (float*)alloc((size_t)N * 64 * 4);
  float* hA   = (float*)alloc((size_t)N * 64 * 4);
  float* hB   = (float*)d_out;   // ping-pong partner; overwritten by final softmax

  // weights -> bf16 transposed (L2-resident, reused by every tile)
  cvt_transpose_bf16<<<(256 * 512 + 255) / 256, 256, 0, stream>>>(W1, W1T, 512, 256);
  cvt_transpose_bf16<<<(64 * 256 + 255) / 256, 256, 0, stream>>>(W2, W2T, 256, 64);

  // symmetric GCN normalization
  zero_f32<<<(N + 255) / 256, 256, 0, stream>>>(dinv, N);
  deg_accum<<<(E + 255) / 256, 256, 0, stream>>>(dst, dinv, E);
  deg_finalize<<<(N + 255) / 256, 256, 0, stream>>>(dinv, N);
  edge_norm<<<(E + 255) / 256, 256, 0, stream>>>(src, dst, dinv, norm, E);

  // fused MLP (WMMA bf16) -> h0 and initial h
  mlp_wmma<<<N / 16, 128, 0, stream>>>(x, W1T, b1, W2T, b2, h0, hA);

  // 10 APPNP steps, ping-pong; 10 is even so the final h lands back in hA
  float* cur = hA;
  float* nxt = hB;
  const long long nvec4 = (long long)N * 16;
  for (int k = 0; k < 10; ++k) {
    prop_init<<<(int)((nvec4 + 255) / 256), 256, 0, stream>>>(cur, h0, dinv, nxt, nvec4);
    prop_edges<<<4096, 256, 0, stream>>>(src, dst, norm, cur, nxt, E);
    float* t = cur; cur = nxt; nxt = t;
  }

  log_softmax64<<<(N + 7) / 8, 256, 0, stream>>>(cur, (float*)d_out, N);
}
